// BinConv2d_6949257085406
// MI455X (gfx1250) — matrix operand order repaired and checked by reading, not compiled, not run on hardware
//
#include <hip/hip_runtime.h>

typedef _Float16 half1;
typedef __attribute__((ext_vector_type(8)))  _Float16 v8h;
typedef __attribute__((ext_vector_type(16))) _Float16 v16h;
typedef __attribute__((ext_vector_type(8)))  float    v8f;
typedef __attribute__((ext_vector_type(4)))  float    v4f;
typedef __attribute__((ext_vector_type(4)))  int      v4i;

#define N_IMG  16
#define C_IN   64
#define C_OUT  64
#define H_IMG  224
#define W_IMG  224
#define HW     (H_IMG * W_IMG)        // 50176
#define NPIX   (N_IMG * HW)           // 802816
#define BN_EPS 1e-4f

// ---------------------------------------------------------------------------
// Kernel 1: per-channel batch statistics -> scale/shift for fused BN
// scale[c] = gamma[c] * rsqrt(var + eps);  shift[c] = beta[c] - mean*scale[c]
// ---------------------------------------------------------------------------
__global__ __launch_bounds__(256) void bn_stats_kernel(
    const float* __restrict__ x, const float* __restrict__ bnw,
    const float* __restrict__ bnb, float* __restrict__ scale,
    float* __restrict__ shift) {
  __shared__ float ssum[256];
  __shared__ float ssq[256];
  const int c = blockIdx.x;
  const int tid = threadIdx.x;
  float s = 0.f, q = 0.f;
  for (int n = 0; n < N_IMG; ++n) {
    const float* p = x + (size_t)(n * C_IN + c) * HW;
    for (int i = tid; i < HW; i += 256) {
      float v = p[i];
      s += v;
      q += v * v;
    }
  }
  ssum[tid] = s;
  ssq[tid] = q;
  __syncthreads();
  for (int off = 128; off > 0; off >>= 1) {
    if (tid < off) {
      ssum[tid] += ssum[tid + off];
      ssq[tid]  += ssq[tid + off];
    }
    __syncthreads();
  }
  if (tid == 0) {
    const float inv = 1.0f / (float)(N_IMG * HW);
    const float mean = ssum[0] * inv;
    const float var  = ssq[0] * inv - mean * mean;
    const float sc   = bnw[c] * rsqrtf(var + BN_EPS);
    scale[c] = sc;
    shift[c] = bnb[c] - mean * sc;
  }
}

// ---------------------------------------------------------------------------
// Kernel 2: fused BN + sign(), NCHW f32 -> NHWC f16 via LDS transpose tile.
// Tile: 32 pixels x 64 channels. Reads coalesced along W, writes coalesced
// along C (128B per pixel).
// ---------------------------------------------------------------------------
__global__ __launch_bounds__(256) void binarize_kernel(
    const float* __restrict__ x, const float* __restrict__ scale,
    const float* __restrict__ shift, half1* __restrict__ xb) {
  __shared__ half1 tile[32 * 64];
  __shared__ float s_sc[64];
  __shared__ float s_sh[64];
  const int tid = threadIdx.x;
  if (tid < 64) {
    s_sc[tid] = scale[tid];
    s_sh[tid] = shift[tid];
  }
  __syncthreads();

  const int b = blockIdx.x;            // 16 * 224 * 7 blocks
  const int wblk = b % 7;
  const int h = (b / 7) % H_IMG;
  const int n = b / (7 * H_IMG);
  const int w0 = wblk * 32;

  // read phase: lane = w, tid/32 = channel sub-slot (8 channels per pass)
  const int wlane = tid & 31;
  const int cbase = tid >> 5;          // 0..7
#pragma unroll
  for (int it = 0; it < 8; ++it) {
    const int c = it * 8 + cbase;
    const float v = x[(size_t)(n * C_IN + c) * HW + h * W_IMG + w0 + wlane];
    const float t = s_sc[c] * v + s_sh[c];
    const half1 sgn =
        t > 0.f ? (half1)1.0f : (t < 0.f ? (half1)-1.0f : (half1)0.0f);
    tile[wlane * 64 + c] = sgn;
  }
  __syncthreads();

  // write phase: tid%64 = channel (contiguous 128B per pixel), tid/64 = pixel
  const int c = tid & 63;
  const int wp = tid >> 6;             // 0..3
#pragma unroll
  for (int it = 0; it < 8; ++it) {
    const int w = it * 4 + wp;
    xb[((size_t)((n * H_IMG + h) * W_IMG) + w0 + w) * 64 + c] =
        tile[w * 64 + c];
  }
}

// ---------------------------------------------------------------------------
// Kernel 3: pack conv weights (OIHW f32) into per-lane WMMA B-fragment layout
// wpack[tap][kchunk][ntile][lane][j] (f16), j<8: K 8h+j, j>=8: K 16+8h+(j-8).
// B 32x16: lanes 0-15 hold K 0..7,16..23 of column n=lane; lanes 16-31 hold K 8..15,24..31.
// ---------------------------------------------------------------------------
__global__ __launch_bounds__(256) void pack_w_kernel(
    const float* __restrict__ wgt, half1* __restrict__ wpack) {
  const int idx = blockIdx.x * 256 + threadIdx.x;   // < 36864
  const int j   = idx & 15;
  const int l   = (idx >> 4) & 31;
  const int nt  = (idx >> 9) & 3;
  const int cch = (idx >> 11) & 1;
  const int tap = idx >> 12;                        // 0..8
  const int kloc = ((j >> 3) << 4) + ((l >> 4) << 3) + (j & 7); // 0..31
  const int cin  = cch * 32 + kloc;                 // 0..63
  const int cout = nt * 16 + (l & 15);
  const int ky = tap / 3, kx = tap % 3;
  wpack[idx] = (half1)wgt[((cout * C_IN + cin) * 3 + ky) * 3 + kx];
}

// ---------------------------------------------------------------------------
// Kernel 4: 3x3 conv as implicit GEMM via v_wmma_f32_16x16x32_f16.
// 8 waves/block; each wave: one 16-pixel M tile x 64 couts (4 N tiles),
// K = 9 taps * 64 cin = 576 = 18 wmma k-steps. B fragments staged in LDS
// via gfx1250 async global->LDS DMA (ASYNCcnt path). A fragments loaded
// straight from NHWC f16 global (L2-resident, 103MB < 192MB L2).
// ---------------------------------------------------------------------------
__global__ __launch_bounds__(256) void conv_wmma_kernel(
    const half1* __restrict__ xb, const half1* __restrict__ wpack,
    const float* __restrict__ bias, float* __restrict__ out) {
  extern __shared__ char smem[];                    // 73728 B = packed weights

#if __has_builtin(__builtin_amdgcn_global_load_async_to_lds_b128)
  {
    typedef __attribute__((address_space(1))) v4i gv4i;
    typedef __attribute__((address_space(3))) v4i lv4i;
    gv4i* gp = (gv4i*)(v4i*)const_cast<half1*>(wpack);
    lv4i* lp = (lv4i*)(v4i*)smem;
    gp += threadIdx.x;
    lp += threadIdx.x;
#pragma unroll
    for (int i = 0; i < 18; ++i) {                  // 18 * 4096B = 73728B
      __builtin_amdgcn_global_load_async_to_lds_b128(gp + i * 256,
                                                     lp + i * 256, 0, 0);
    }
#if __has_builtin(__builtin_amdgcn_s_wait_asynccnt)
    __builtin_amdgcn_s_wait_asynccnt(0);
#else
    asm volatile("s_wait_asynccnt 0x0" ::: "memory");
#endif
  }
#else
  {
    const uint4* src = (const uint4*)wpack;
    uint4* dst = (uint4*)smem;
    for (int i = threadIdx.x; i < 4608; i += 256) dst[i] = src[i];
  }
#endif
  __syncthreads();
  const half1* wl = (const half1*)smem;

  const int wave = threadIdx.x >> 5;
  const int lane = threadIdx.x & 31;
  const int t = blockIdx.x * 8 + wave;              // 50176 tiles total
  const int pixbase = t << 4;
  const int n = pixbase / HW;
  const int rem = pixbase - n * HW;
  const int h = rem / W_IMG;
  const int w0 = rem - h * W_IMG;                   // multiple of 16

  const int m = lane & 15;                          // pixel row in tile
  const bool hiK = lane >= 16;                      // A-layout K half-select

  v8f acc0 = {}, acc1 = {}, acc2 = {}, acc3 = {};
  const v8h vz = {};

  for (int ky = 0; ky < 3; ++ky) {
    const int hs = h + ky - 1;
    const bool vrow = (unsigned)hs < (unsigned)H_IMG;
    for (int kx = 0; kx < 3; ++kx) {
      const int tap = ky * 3 + kx;
      const int wsx = w0 + m + kx - 1;
      const bool valid = vrow && ((unsigned)wsx < (unsigned)W_IMG);
      // clamp address in-bounds; fragment zeroed when invalid (conv padding)
      const int P = (n * H_IMG + (vrow ? hs : 0)) * W_IMG + (valid ? wsx : 0);
      const half1* ap = xb + (size_t)P * 64;
#pragma unroll
      for (int c = 0; c < 2; ++c) {
        const int off = c * 32 + (hiK ? 8 : 0);
        v8h alo = *(const v8h*)(ap + off);          // K off..off+7
        v8h ahi = *(const v8h*)(ap + off + 16);     // K off+16..off+23
        alo = valid ? alo : vz;
        ahi = valid ? ahi : vz;
        const v16h a = __builtin_shufflevector(
            alo, ahi, 0, 1, 2, 3, 4, 5, 6, 7, 8, 9, 10, 11, 12, 13, 14, 15);
        const half1* bbase = wl + ((((tap * 2 + c) * 4) * 32) + lane) * 16;
        const v16h b0 = *(const v16h*)(bbase);
        acc0 = __builtin_amdgcn_wmma_f32_16x16x32_f16(false, a, false, b0,
                                                      (short)0, acc0, false,
                                                      false);
        const v16h b1 = *(const v16h*)(bbase + 512);
        acc1 = __builtin_amdgcn_wmma_f32_16x16x32_f16(false, a, false, b1,
                                                      (short)0, acc1, false,
                                                      false);
        const v16h b2 = *(const v16h*)(bbase + 1024);
        acc2 = __builtin_amdgcn_wmma_f32_16x16x32_f16(false, a, false, b2,
                                                      (short)0, acc2, false,
                                                      false);
        const v16h b3 = *(const v16h*)(bbase + 1536);
        acc3 = __builtin_amdgcn_wmma_f32_16x16x32_f16(false, a, false, b3,
                                                      (short)0, acc3, false,
                                                      false);
      }
    }
  }

  // Store: lane holds 8 consecutive pixels (rows M) of one cout plane.
  const int mbase = hiK ? 8 : 0;
  const int hwoff = h * W_IMG + w0 + mbase;

#define STORE_TILE(NT, ACC)                                              \
  {                                                                      \
    const int cout = (NT) * 16 + m;                                      \
    const float bv = bias[cout];                                         \
    float* op = out + (size_t)(n * C_OUT + cout) * HW + hwoff;           \
    v4f lo = {ACC[0] + bv, ACC[1] + bv, ACC[2] + bv, ACC[3] + bv};       \
    v4f hi = {ACC[4] + bv, ACC[5] + bv, ACC[6] + bv, ACC[7] + bv};       \
    *(v4f*)op = lo;                                                      \
    *(v4f*)(op + 4) = hi;                                                \
  }
  STORE_TILE(0, acc0)
  STORE_TILE(1, acc1)
  STORE_TILE(2, acc2)
  STORE_TILE(3, acc3)
#undef STORE_TILE
}

// ---------------------------------------------------------------------------
extern "C" void kernel_launch(void* const* d_in, const int* in_sizes, int n_in,
                              void* d_out, int out_size, void* d_ws,
                              size_t ws_size, hipStream_t stream) {
  const float* x   = (const float*)d_in[0];
  const float* bnw = (const float*)d_in[1];
  const float* bnb = (const float*)d_in[2];
  const float* cw  = (const float*)d_in[3];
  const float* cb  = (const float*)d_in[4];
  float* out = (float*)d_out;

  char* ws = (char*)d_ws;
  float* scale  = (float*)ws;                  // 64 f32
  float* shift  = (float*)(ws + 256);          // 64 f32
  half1* wpack  = (half1*)(ws + 1024);         // 36864 f16 = 73728 B
  half1* xb     = (half1*)(ws + (1 << 17));    // NPIX*64 f16 = ~103 MB (NHWC)

  bn_stats_kernel<<<C_IN, 256, 0, stream>>>(x, bnw, bnb, scale, shift);
  binarize_kernel<<<N_IMG * H_IMG * 7, 256, 0, stream>>>(x, scale, shift, xb);
  pack_w_kernel<<<144, 256, 0, stream>>>(cw, wpack);
  conv_wmma_kernel<<<NPIX / 16 / 8, 256, 73728, stream>>>(xb, wpack, cb, out);
}
